// NMTDecoder_55817394979255
// MI455X (gfx1250) — matrix operand order repaired and checked by reading, not compiled, not run on hardware
//
#include <hip/hip_runtime.h>
#include <hip/hip_bf16.h>

// ---------------- problem constants ----------------
constexpr int V  = 32000;   // vocab
constexpr int E  = 256;     // embedding
constexpr int H  = 512;     // hidden
constexpr int B  = 64;      // batch
constexpr int S  = 64;      // src len
constexpr int T  = 64;      // tgt len
constexpr int KIN = E + H;  // 768  GRU input width
constexpr int C2  = 2 * H;  // 1024 classifier K

typedef __attribute__((ext_vector_type(16))) __bf16        v16bf;
typedef __attribute__((ext_vector_type(8)))  float         v8f;
typedef __attribute__((ext_vector_type(4)))  float         v4f;
typedef __attribute__((ext_vector_type(8)))  unsigned int  v8u;
typedef __attribute__((ext_vector_type(4)))  unsigned int  v4u;

// float -> bf16 bits, round-to-nearest-even
__device__ __forceinline__ unsigned short f2bf(float x) {
    unsigned int u = __builtin_bit_cast(unsigned int, x);
    return (unsigned short)((u + 0x7FFFu + ((u >> 16) & 1u)) >> 16);
}
__device__ __forceinline__ unsigned int pack2(float lo, float hi) {
    return (unsigned int)f2bf(lo) | ((unsigned int)f2bf(hi) << 16);
}
__device__ __forceinline__ v8f wmma_bf16(v16bf a, v16bf b, v8f c) {
    return __builtin_amdgcn_wmma_f32_16x16x32_bf16(false, a, false, b, (short)0, c, false, false);
}
// Build a v16bf fragment from two 16-byte chunks (per documented CDNA5 layouts both the
// A fragment (K0-7|K16-23 per lane-half) and B fragment (same K0-7|K16-23 order)
// are exactly two contiguous 16B runs per lane).
__device__ __forceinline__ v16bf ld_frag2(const unsigned short* p0, const unsigned short* p1) {
    v4u lo = *(const v4u*)p0;
    v4u hi = *(const v4u*)p1;
    v8u u;
    u[0] = lo[0]; u[1] = lo[1]; u[2] = lo[2]; u[3] = lo[3];
    u[4] = hi[0]; u[5] = hi[1]; u[6] = hi[2]; u[7] = hi[3];
    return __builtin_bit_cast(v16bf, u);
}
__device__ __forceinline__ float sigmoidf_(float x) { return 1.0f / (1.0f + __expf(-x)); }

// ---------------- generic f32 -> bf16 converter (NT loads: sources read once) ----------
__global__ __launch_bounds__(256) void k_cvt(const float* __restrict__ src,
                                             unsigned short* __restrict__ dst, int n) {
    int i = (blockIdx.x * 256 + threadIdx.x) * 4;
    if (i >= n) return;
    v4f f = __builtin_nontemporal_load((const v4f*)(src + i));
    *(unsigned int*)(dst + i)     = pack2(f[0], f[1]);
    *(unsigned int*)(dst + i + 2) = pack2(f[2], f[3]);
}

// ---------------- h0 = init_hidden @ W_map^T + b_map ; ctx0 = 0 ----------------
__global__ __launch_bounds__(256) void k_init(const float* __restrict__ x,
                                              const float* __restrict__ W_map,
                                              const float* __restrict__ b_map,
                                              float* __restrict__ h_all,
                                              unsigned short* __restrict__ h_bf,
                                              unsigned short* __restrict__ ctx_bf) {
    int gid = blockIdx.x * 256 + threadIdx.x;   // 0 .. B*H-1 (grid = 128)
    int b = gid >> 9, j = gid & 511;
    const float* xr = x + (size_t)b * H;
    const float* wr = W_map + (size_t)j * H;
    float s = b_map[j];
    for (int k = 0; k < H; ++k) s += xr[k] * wr[k];
    h_all[gid]  = s;
    h_bf[gid]   = f2bf(s);
    ctx_bf[gid] = 0;
}

// ---------------- GRU cell, one 16x16 tile of h per wave (grid 4 x 32, 32 thr) ----------
// All WMMA operands are pre-materialized bf16 -> fragments are pure b128 loads.
__global__ __launch_bounds__(32) void k_gru(const int* __restrict__ tseq,
                                            const unsigned short* __restrict__ embb,
                                            const unsigned short* __restrict__ wihb,
                                            const float* __restrict__ b_ih,
                                            const unsigned short* __restrict__ whhb,
                                            const float* __restrict__ b_hh,
                                            const unsigned short* __restrict__ ctx_bf,
                                            const unsigned short* __restrict__ h_bf,
                                            float* __restrict__ h_all,
                                            unsigned short* __restrict__ h_bf_out_base,
                                            int t) {
    const int lane = threadIdx.x;
    const int half = lane >> 4;
    const int m    = lane & 15;          // A row within tile / B column within tile
    const int b0   = blockIdx.x * 16;
    const int j0   = blockIdx.y * 16;
    const int brow = b0 + m;

    const int yidx = tseq[brow * T + t];
    const unsigned short* embrow = embb   + (size_t)yidx * E;
    const unsigned short* ctxrow = ctx_bf + ((size_t)t * B + brow) * H;
    const unsigned short* hrow   = h_bf   + ((size_t)t * B + brow) * H;

    const unsigned short* wr = wihb + (size_t)(        j0 + m) * KIN;
    const unsigned short* wz = wihb + (size_t)(H     + j0 + m) * KIN;
    const unsigned short* wn = wihb + (size_t)(2 * H + j0 + m) * KIN;
    const unsigned short* ur = whhb + (size_t)(        j0 + m) * H;
    const unsigned short* uz = whhb + (size_t)(H     + j0 + m) * H;
    const unsigned short* un = whhb + (size_t)(2 * H + j0 + m) * H;

    v8f aIr = {}, aIz = {}, aIn = {}, aHr = {}, aHz = {}, aHn = {};

    // gi = rnn_in @ W_ih^T : K = 768 (emb[0:256] ++ ctx[0:512])
    for (int k0 = 0; k0 < KIN; k0 += 32) {
        const unsigned short* src = (k0 < E) ? (embrow + k0) : (ctxrow + (k0 - E));
        v16bf a = ld_frag2(src + 8 * half, src + 16 + 8 * half);
        const int kb = k0 + 8 * half;
        aIr = wmma_bf16(a, ld_frag2(wr + kb, wr + kb + 16), aIr);
        aIz = wmma_bf16(a, ld_frag2(wz + kb, wz + kb + 16), aIz);
        aIn = wmma_bf16(a, ld_frag2(wn + kb, wn + kb + 16), aIn);
    }
    // gh = h @ W_hh^T : K = 512
    for (int k0 = 0; k0 < H; k0 += 32) {
        const unsigned short* src = hrow + k0;
        v16bf a = ld_frag2(src + 8 * half, src + 16 + 8 * half);
        const int kb = k0 + 8 * half;
        aHr = wmma_bf16(a, ld_frag2(ur + kb, ur + kb + 16), aHr);
        aHz = wmma_bf16(a, ld_frag2(uz + kb, uz + kb + 16), aHz);
        aHn = wmma_bf16(a, ld_frag2(un + kb, un + kb + 16), aHn);
    }

    // gates + state update; C/D element (v): row = v + 8*half, col = m
    const int j = j0 + m;
    const float bir = b_ih[j], biz = b_ih[H + j], bin = b_ih[2 * H + j];
    const float bhr = b_hh[j], bhz = b_hh[H + j], bhn = b_hh[2 * H + j];
    const float*    hin   = h_all + (size_t)t * B * H;
    float*          hout  = h_all + (size_t)(t + 1) * B * H;
    unsigned short* hbout = h_bf_out_base + (size_t)(t + 1) * B * H;
#pragma unroll
    for (int v = 0; v < 8; ++v) {
        int bm = b0 + v + 8 * half;
        float r  = sigmoidf_(aIr[v] + bir + aHr[v] + bhr);
        float z  = sigmoidf_(aIz[v] + biz + aHz[v] + bhz);
        float nn = tanhf(aIn[v] + bin + r * (aHn[v] + bhn));
        float hp = hin[(size_t)bm * H + j];
        float hv = (1.0f - z) * nn + z * hp;
        hout[(size_t)bm * H + j]  = hv;
        hbout[(size_t)bm * H + j] = f2bf(hv);
    }
}

// ---------------- attention + context + pred-row emit (one block per batch row) ---------
__global__ __launch_bounds__(256) void k_attn(const float* __restrict__ enc,
                                              const float* __restrict__ h_all,
                                              unsigned short* __restrict__ ctx_bf,
                                              unsigned short* __restrict__ pred,
                                              int t) {
    __shared__ float sh[H];
    __shared__ float sc[S];
    const int b   = blockIdx.x;
    const int tid = threadIdx.x;
    const int wid = tid >> 5, lane = tid & 31;

    const float* hrow = h_all + ((size_t)(t + 1) * B + b) * H;
    sh[tid]       = hrow[tid];
    sh[tid + 256] = hrow[tid + 256];
    __syncthreads();

    // scores[s] = enc[b,s,:] . h   (each wave: 8 values of s)
    for (int si = 0; si < 8; ++si) {
        int s = wid * 8 + si;
        const float* er = enc + ((size_t)b * S + s) * H;
        float sum = 0.0f;
        for (int k = lane; k < H; k += 32) sum += er[k] * sh[k];
#pragma unroll
        for (int off = 16; off > 0; off >>= 1) sum += __shfl_down(sum, off, 32);
        if (lane == 0) sc[s] = sum;
    }
    __syncthreads();

    // softmax over S=64, wave 0 (2 values per lane)
    if (wid == 0) {
        float x0 = sc[lane], x1 = sc[lane + 32];
        float mx = fmaxf(x0, x1);
#pragma unroll
        for (int off = 16; off > 0; off >>= 1) mx = fmaxf(mx, __shfl_xor(mx, off, 32));
        float e0 = __expf(x0 - mx), e1 = __expf(x1 - mx);
        float sm = e0 + e1;
#pragma unroll
        for (int off = 16; off > 0; off >>= 1) sm += __shfl_xor(sm, off, 32);
        float inv = 1.0f / sm;
        sc[lane] = e0 * inv; sc[lane + 32] = e1 * inv;
    }
    __syncthreads();

    // ctx[j] = sum_s p[s] * enc[b,s,j]  (each thread: j, j+256)
    float c0 = 0.0f, c1 = 0.0f;
    for (int s = 0; s < S; ++s) {
        float p = sc[s];
        const float* er = enc + ((size_t)b * S + s) * H;
        c0 += p * er[tid];
        c1 += p * er[tid + 256];
    }
    unsigned short* cb = ctx_bf + ((size_t)(t + 1) * B + b) * H;
    cb[tid]       = f2bf(c0);
    cb[tid + 256] = f2bf(c1);

    unsigned short* pr = pred + ((size_t)t * B + b) * C2;  // [ctx | h] bf16 row
    pr[tid]           = f2bf(c0);
    pr[tid + 256]     = f2bf(c1);
    pr[H + tid]       = f2bf(sh[tid]);
    pr[H + tid + 256] = f2bf(sh[tid + 256]);
}

// ---------------- big classifier GEMM: [T*B,1024] x [1024,V] -> out[B,T,V] -------------
constexpr int BM  = 128;
constexpr int BN  = 128;
constexpr int BK  = 32;
constexpr int LDP = 40;   // LDS row stride in bf16: rows 16B-aligned, conflict-free gathers

__device__ __forceinline__ v16bf lds_frag_a(const unsigned short* sA, int m0, int lane) {
    const unsigned short* row = sA + (m0 + (lane & 15)) * LDP;
    int half = lane >> 4;
    return ld_frag2(row + 8 * half, row + 16 + 8 * half);
}
__device__ __forceinline__ v16bf lds_frag_b(const unsigned short* sB, int n0, int lane) {
    const unsigned short* row = sB + (n0 + (lane & 15)) * LDP;
    int half = lane >> 4;
    return ld_frag2(row + 8 * half, row + 16 + 8 * half);
}

__global__ __launch_bounds__(256) void k_gemm(const unsigned short* __restrict__ Ap,  // [T*B, C2] bf16
                                              const unsigned short* __restrict__ Bw,  // [V,  C2] bf16
                                              const float* __restrict__ bias,         // [V]
                                              float* __restrict__ out) {              // [B,T,V]
    __shared__ unsigned short sA[BM * LDP];
    __shared__ unsigned short sB[BN * LDP];
    const int tid  = threadIdx.x;
    const int lane = tid & 31, wid = tid >> 5;
    const int c0 = blockIdx.x * BN;     // 250 tiles cover V exactly
    const int r0 = blockIdx.y * BM;     // 32 tiles cover T*B exactly
    const int wm = wid >> 2;            // 0..1 -> 64-row slab
    const int wn = wid & 3;             // 0..3 -> 32-col slab

    v8f acc[4][2];
#pragma unroll
    for (int i = 0; i < 4; ++i)
#pragma unroll
        for (int jj = 0; jj < 2; ++jj) acc[i][jj] = v8f{};

    for (int kt = 0; kt < C2 / BK; ++kt) {
        const int k0 = kt * BK;
        // stage 128x32 A tile and 128x32 B tile as b128 chunks (2 per thread per tile)
#pragma unroll
        for (int i = 0; i < 2; ++i) {
            int idx = tid + i * 256;          // 0..511 chunk id
            int row = idx >> 2;
            int ch  = (idx & 3) * 8;          // ushort offset within row
            *(v4u*)(sA + row * LDP + ch) = *(const v4u*)(Ap + (size_t)(r0 + row) * C2 + k0 + ch);
            *(v4u*)(sB + row * LDP + ch) = *(const v4u*)(Bw + (size_t)(c0 + row) * C2 + k0 + ch);
        }
        __syncthreads();

        v16bf af[4], bfm[2];
#pragma unroll
        for (int i = 0; i < 4; ++i)   af[i]   = lds_frag_a(sA, wm * 64 + i * 16, lane);
#pragma unroll
        for (int jj = 0; jj < 2; ++jj) bfm[jj] = lds_frag_b(sB, wn * 32 + jj * 16, lane);
#pragma unroll
        for (int i = 0; i < 4; ++i)
#pragma unroll
            for (int jj = 0; jj < 2; ++jj) acc[i][jj] = wmma_bf16(af[i], bfm[jj], acc[i][jj]);
        __syncthreads();
    }

    // epilogue: +bias, NT-scatter to out[b, t, v] (write-once stream; keep W_cls in L2)
    const int half = lane >> 4, n = lane & 15;
#pragma unroll
    for (int i = 0; i < 4; ++i) {
#pragma unroll
        for (int jj = 0; jj < 2; ++jj) {
            int c = c0 + wn * 32 + jj * 16 + n;
            float bb = bias[c];
#pragma unroll
            for (int v = 0; v < 8; ++v) {
                int r   = r0 + wm * 64 + i * 16 + v + 8 * half;
                int tt  = r >> 6;         // / B
                int bro = r & 63;         // % B
                __builtin_nontemporal_store(acc[i][jj][v] + bb,
                                            out + ((size_t)bro * T + tt) * V + c);
            }
        }
    }
}

// ---------------- host launcher ----------------
extern "C" void kernel_launch(void* const* d_in, const int* in_sizes, int n_in,
                              void* d_out, int out_size, void* d_ws, size_t ws_size,
                              hipStream_t stream) {
    const float* enc   = (const float*)d_in[0];
    const float* h0in  = (const float*)d_in[1];
    const int*   tseq  = (const int*)  d_in[2];
    const float* emb   = (const float*)d_in[3];
    const float* W_ih  = (const float*)d_in[4];
    const float* b_ih  = (const float*)d_in[5];
    const float* W_hh  = (const float*)d_in[6];
    const float* b_hh  = (const float*)d_in[7];
    const float* W_map = (const float*)d_in[8];
    const float* b_map = (const float*)d_in[9];
    const float* W_cls = (const float*)d_in[10];
    const float* b_cls = (const float*)d_in[11];
    float* out = (float*)d_out;

    char* p = (char*)d_ws;
    float* h_all = (float*)p;                    p += (size_t)(T + 1) * B * H * 4;  // 8.52 MB
    unsigned short* h_bf   = (unsigned short*)p; p += (size_t)(T + 1) * B * H * 2;  // 4.26 MB
    unsigned short* ctx_bf = (unsigned short*)p; p += (size_t)(T + 1) * B * H * 2;  // 4.26 MB
    unsigned short* predb  = (unsigned short*)p; p += (size_t)T * B * C2 * 2;       // 8.39 MB
    unsigned short* wclsb  = (unsigned short*)p; p += (size_t)V * C2 * 2;           // 65.5 MB
    unsigned short* embb   = (unsigned short*)p; p += (size_t)V * E * 2;            // 16.4 MB
    unsigned short* wihb   = (unsigned short*)p; p += (size_t)3 * H * KIN * 2;      // 2.36 MB
    unsigned short* whhb   = (unsigned short*)p;                                    // 1.57 MB

    auto cvt = [&](const float* s, unsigned short* d, int n) {
        k_cvt<<<dim3((n / 4 + 255) / 256), dim3(256), 0, stream>>>(s, d, n);
    };
    // 1) one-shot bf16 materialization of all reused WMMA operands
    cvt(W_cls, wclsb, V * C2);          // stays L2-resident for the big GEMM
    cvt(emb,   embb,  V * E);
    cvt(W_ih,  wihb,  3 * H * KIN);
    cvt(W_hh,  whhb,  3 * H * H);
    // 2) h0 = x @ W_map^T + b_map ; ctx0 = 0
    k_init<<<dim3((B * H) / 256), dim3(256), 0, stream>>>(h0in, W_map, b_map, h_all, h_bf, ctx_bf);
    // 3) sequential recurrence: GRU (WMMA, pure b128 fragment loads) + attention
    for (int t = 0; t < T; ++t) {
        k_gru<<<dim3(B / 16, H / 16), dim3(32), 0, stream>>>(tseq, embb, wihb, b_ih, whhb, b_hh,
                                                             ctx_bf, h_bf, h_all, h_bf, t);
        k_attn<<<dim3(B), dim3(256), 0, stream>>>(enc, h_all, ctx_bf, predb, t);
    }
    // 4) one big WMMA GEMM for all T*B logits rows
    k_gemm<<<dim3(V / BN, (T * B) / BM), dim3(256), 0, stream>>>(predb, wclsb, b_cls, out);
}
